// SpaSeqNetLast_35476429865423
// MI455X (gfx1250) — matrix operand order repaired and checked by reading, not compiled, not run on hardware
//
#include <hip/hip_runtime.h>
#include <hip/hip_bf16.h>

// Problem constants (match reference)
#define TT  3
#define NN  16384
#define EE  262144
#define HH  512   // F == H == 512

typedef __attribute__((ext_vector_type(16))) __bf16 v16bf;
typedef __attribute__((ext_vector_type(8)))  float  v8f;

__device__ __forceinline__ unsigned short f2bf(float f) {
    unsigned int u = __float_as_uint(f);
    unsigned int r = u + 0x7FFFu + ((u >> 16) & 1u);   // round-to-nearest-even
    return (unsigned short)(r >> 16);
}

// ---------------------------------------------------------------------------
// Weight pre-pass: W[k][col] f32 -> Bt[col][k] bf16 (transposed, WMMA-B layout)
// ---------------------------------------------------------------------------
__global__ __launch_bounds__(256) void wtrans_kernel(const float* __restrict__ W,
                                                     unsigned short* __restrict__ Bt) {
    int idx = blockIdx.x * 256 + threadIdx.x;     // over 512*512
    int k   = idx >> 9;
    int col = idx & 511;
    Bt[col * HH + k] = f2bf(W[idx]);
}

// Vectorized fp32 -> bf16 conversion (activations), float4 -> ushort4
__global__ __launch_bounds__(256) void cvt_bf16_kernel(const float* __restrict__ in,
                                                       unsigned short* __restrict__ outb,
                                                       int n4) {
    int i = blockIdx.x * 256 + threadIdx.x;
    if (i < n4) {
        float4 v = ((const float4*)in)[i];
        ushort4 o;
        o.x = f2bf(v.x); o.y = f2bf(v.y); o.z = f2bf(v.z); o.w = f2bf(v.w);
        ((ushort4*)outb)[i] = o;
    }
}

// Fused ReLU + fp32 -> bf16 (conv-1 epilogue feeding GEMM3)
__global__ __launch_bounds__(256) void relu_cvt_kernel(const float* __restrict__ in,
                                                       unsigned short* __restrict__ outb,
                                                       int n4) {
    int i = blockIdx.x * 256 + threadIdx.x;
    if (i < n4) {
        float4 v = ((const float4*)in)[i];
        ushort4 o;
        o.x = f2bf(fmaxf(v.x, 0.f)); o.y = f2bf(fmaxf(v.y, 0.f));
        o.z = f2bf(fmaxf(v.z, 0.f)); o.w = f2bf(fmaxf(v.w, 0.f));
        ((ushort4*)outb)[i] = o;
    }
}

// ---------------------------------------------------------------------------
// GEMM: C[M,512] = A[M,512](bf16) * B(bf16, transposed Bt[col][k]) ; f32 accum
// grid = (M/128, 8), block = 256 (8 waves). Wave -> 16 rows x 64 cols.
// Inner loop: 2 A b128 loads + 8 B b128 loads + 4 WMMA (no conversion VALU).
// ---------------------------------------------------------------------------
__device__ __forceinline__ v16bf loadB(const unsigned short* __restrict__ Bt,
                                       int col, int k0, int kbB) {
    union { v16bf v; uint4 q[2]; } b;
    const uint4* p = (const uint4*)(Bt + (size_t)col * HH + k0 + kbB);
    b.q[0] = p[0];
    b.q[1] = p[1];
    return b.v;
}

template <bool OUT_BF16>
__global__ __launch_bounds__(256) void gemm_bf16_wmma_kernel(
    const unsigned short* __restrict__ A,   // bf16 [M,512]
    const unsigned short* __restrict__ Bt,  // bf16 [512,512] transposed
    float* __restrict__ Cf,                 // f32 out (used when !OUT_BF16)
    unsigned short* __restrict__ Cb,        // bf16 out (used when OUT_BF16)
    int M) {
    const int lane = threadIdx.x & 31;
    const int wave = threadIdx.x >> 5;                  // 0..7
    const int rowBase = blockIdx.x * 128 + wave * 16;
    const int colBase = blockIdx.y * 64;
    const int r   = lane & 15;
    const int hi  = lane >> 4;                          // 0 or 1
    const int kbA = hi * 16;                            // A: contiguous 16-K run
    const int kbB = hi * 16;                            // B: contiguous 16-K run

    v8f acc0 = {}, acc1 = {}, acc2 = {}, acc3 = {};

    const unsigned short* arow = A + (size_t)(rowBase + r) * HH;
    const int c0 = colBase + r;

    for (int k0 = 0; k0 < HH; k0 += 32) {
        union { v16bf v; uint4 q[2]; } af;
        af.q[0] = *(const uint4*)(arow + k0 + kbA);        // K = k0+kbA .. +7
        af.q[1] = *(const uint4*)(arow + k0 + 8 + kbA);    // K = k0+8+kbA .. +7

        v16bf b0 = loadB(Bt, c0 +  0, k0, kbB);
        v16bf b1 = loadB(Bt, c0 + 16, k0, kbB);
        v16bf b2 = loadB(Bt, c0 + 32, k0, kbB);
        v16bf b3 = loadB(Bt, c0 + 48, k0, kbB);

        acc0 = __builtin_amdgcn_wmma_f32_16x16x32_bf16(false, af.v, false, b0, (short)0, acc0, false, false);
        acc1 = __builtin_amdgcn_wmma_f32_16x16x32_bf16(false, af.v, false, b1, (short)0, acc1, false, false);
        acc2 = __builtin_amdgcn_wmma_f32_16x16x32_bf16(false, af.v, false, b2, (short)0, acc2, false, false);
        acc3 = __builtin_amdgcn_wmma_f32_16x16x32_bf16(false, af.v, false, b3, (short)0, acc3, false, false);
    }

    // store: v8f element rv -> row rowBase + rv + 8*hi, col c0 + 16*tile
    #pragma unroll
    for (int rv = 0; rv < 8; ++rv) {
        int row = rowBase + rv + (hi << 3);
        if (OUT_BF16) {
            unsigned short* crow = Cb + (size_t)row * HH + c0;
            crow[0]  = f2bf(acc0[rv]);
            crow[16] = f2bf(acc1[rv]);
            crow[32] = f2bf(acc2[rv]);
            crow[48] = f2bf(acc3[rv]);
        } else {
            float* crow = Cf + (size_t)row * HH + c0;
            crow[0]  = acc0[rv];
            crow[16] = acc1[rv];
            crow[32] = acc2[rv];
            crow[48] = acc3[rv];
        }
    }
}

// ---------------------------------------------------------------------------
// Degree / normalization helpers
// ---------------------------------------------------------------------------
__global__ __launch_bounds__(256) void fill_kernel(float* __restrict__ p, float v, int n) {
    int i = blockIdx.x * 256 + threadIdx.x;
    if (i < n) p[i] = v;
}

__global__ __launch_bounds__(256) void deg_kernel(const int* __restrict__ dst,
                                                  float* __restrict__ deg, int E) {
    int e = blockIdx.x * 256 + threadIdx.x;
    if (e < E) atomicAdd(&deg[dst[e]], 1.0f);
}

__global__ __launch_bounds__(256) void rsqrt_kernel(float* __restrict__ p, int n) {
    int i = blockIdx.x * 256 + threadIdx.x;
    if (i < n) p[i] = rsqrtf(p[i]);     // deg >= 1 from self-loop init
}

// out[n][j] = b[j]   (bias is the accumulator init for the scatter-add)
__global__ __launch_bounds__(256) void bias_init_kernel(float* __restrict__ out,
                                                        const float* __restrict__ b, int total) {
    int i = blockIdx.x * 256 + threadIdx.x;
    if (i < total) out[i] = b[i & (HH - 1)];
}

// ---------------------------------------------------------------------------
// Edge scatter-add: out[dst] += m[src] * dinv[src]*dinv[dst]
// 128 threads per edge (4 floats each via float4), 2 edges per 256-thread block
// ---------------------------------------------------------------------------
__global__ __launch_bounds__(256) void agg_kernel(
    const float* __restrict__ m, const int* __restrict__ src, const int* __restrict__ dst,
    const float* __restrict__ dinv, float* __restrict__ out, int E) {
    int sub = threadIdx.x & 127;
    int e   = blockIdx.x * 2 + (threadIdx.x >> 7);
    if (e >= E) return;
    int s = src[e], d = dst[e];
    float w = dinv[s] * dinv[d];
    float4 v = ((const float4*)(m + (size_t)s * HH))[sub];
    float* o = out + (size_t)d * HH + sub * 4;
    atomicAdd(o + 0, v.x * w);
    atomicAdd(o + 1, v.y * w);
    atomicAdd(o + 2, v.z * w);
    atomicAdd(o + 3, v.w * w);
}

// Self-loop edge: out[n] += m[n] * dinv[n]^2  (one writer per element, no atomics)
__global__ __launch_bounds__(256) void selfloop_kernel(
    const float* __restrict__ m, const float* __restrict__ dinv,
    float* __restrict__ out, int total) {
    int i = blockIdx.x * 256 + threadIdx.x;
    if (i < total) {
        float dv = dinv[i >> 9];
        out[i] += m[i] * dv * dv;
    }
}

// ---------------------------------------------------------------------------
// Orchestration
// ---------------------------------------------------------------------------
extern "C" void kernel_launch(void* const* d_in, const int* in_sizes, int n_in,
                              void* d_out, int out_size, void* d_ws, size_t ws_size,
                              hipStream_t stream) {
    const int*   e_idx = (const int*)d_in[0];   // [T,2,E] int32
    const float* x     = (const float*)d_in[1]; // [T,N,512]
    const float* Wlin  = (const float*)d_in[2];
    const float* W1    = (const float*)d_in[3];
    const float* b1    = (const float*)d_in[4];
    const float* W2    = (const float*)d_in[5];
    const float* b2    = (const float*)d_in[6];
    float* out = (float*)d_out;                 // [T,N,512]

    // Workspace layout
    unsigned short* WlinT = (unsigned short*)d_ws;           // 512*512 bf16
    unsigned short* W1T   = WlinT + HH * HH;
    unsigned short* W2T   = W1T + HH * HH;
    float* dinv = (float*)(W2T + HH * HH);                   // [T,N] f32
    unsigned short* abuf = (unsigned short*)(dinv + TT * NN);   // [N,512] bf16
    unsigned short* hbf  = abuf + (size_t)NN * HH;              // [N,512] bf16
    float* hbuf = (float*)(hbf + (size_t)NN * HH);              // [N,512] f32
    float* mbuf = hbuf + (size_t)NN * HH;                       // [N,512] f32

    const int WB  = (HH * HH) / 256;           // 1024 blocks for 512x512
    const int NB  = (NN * HH) / 256;           // 32768 blocks for N*512
    const int NB4 = (NN * HH / 4) / 256;       // 8192 blocks, float4-wide
    const dim3 GG(NN / 128, HH / 64);          // GEMM grid: (128, 8)

    // Weight conversion/transpose (once)
    wtrans_kernel<<<WB, 256, 0, stream>>>(Wlin, WlinT);
    wtrans_kernel<<<WB, 256, 0, stream>>>(W1, W1T);
    wtrans_kernel<<<WB, 256, 0, stream>>>(W2, W2T);

    // Degrees: init to 1 (self-loop), count 2E incoming edges per t, rsqrt
    fill_kernel<<<(TT * NN) / 256, 256, 0, stream>>>(dinv, 1.0f, TT * NN);
    for (int t = 0; t < TT; ++t) {
        const int* dstT = e_idx + ((size_t)t * 2 + 1) * EE;
        const int* dstL = e_idx + ((size_t)(TT - 1) * 2 + 1) * EE;
        deg_kernel<<<EE / 256, 256, 0, stream>>>(dstT, dinv + t * NN, EE);
        deg_kernel<<<EE / 256, 256, 0, stream>>>(dstL, dinv + t * NN, EE);
    }
    rsqrt_kernel<<<(TT * NN) / 256, 256, 0, stream>>>(dinv, TT * NN);

    for (int t = 0; t < TT; ++t) {
        const int* srcT = e_idx + ((size_t)t * 2 + 0) * EE;
        const int* dstT = e_idx + ((size_t)t * 2 + 1) * EE;
        const int* srcL = e_idx + ((size_t)(TT - 1) * 2 + 0) * EE;
        const int* dstL = e_idx + ((size_t)(TT - 1) * 2 + 1) * EE;
        const float* dv = dinv + t * NN;

        // abuf = bf16(x[t])
        cvt_bf16_kernel<<<NB4, 256, 0, stream>>>(x + (size_t)t * NN * HH, abuf, NN * HH / 4);
        // hbf = bf16(abuf @ W_lin)   (h only feeds the next GEMM)
        gemm_bf16_wmma_kernel<true><<<GG, 256, 0, stream>>>(abuf, WlinT, (float*)nullptr, hbf, NN);
        // mbuf = f32(hbf @ W1)
        gemm_bf16_wmma_kernel<false><<<GG, 256, 0, stream>>>(hbf, W1T, mbuf, (unsigned short*)nullptr, NN);
        // conv1: hbuf = Agg(mbuf) + b1
        bias_init_kernel<<<NB, 256, 0, stream>>>(hbuf, b1, NN * HH);
        agg_kernel<<<EE / 2, 256, 0, stream>>>(mbuf, srcT, dstT, dv, hbuf, EE);
        agg_kernel<<<EE / 2, 256, 0, stream>>>(mbuf, srcL, dstL, dv, hbuf, EE);
        selfloop_kernel<<<NB, 256, 0, stream>>>(mbuf, dv, hbuf, NN * HH);
        // abuf = bf16(relu(hbuf))    (feeds GEMM3)
        relu_cvt_kernel<<<NB4, 256, 0, stream>>>(hbuf, abuf, NN * HH / 4);
        // mbuf = f32(abuf @ W2)
        gemm_bf16_wmma_kernel<false><<<GG, 256, 0, stream>>>(abuf, W2T, mbuf, (unsigned short*)nullptr, NN);
        // conv2: out[t] = Agg(mbuf) + b2
        float* ot = out + (size_t)t * NN * HH;
        bias_init_kernel<<<NB, 256, 0, stream>>>(ot, b2, NN * HH);
        agg_kernel<<<EE / 2, 256, 0, stream>>>(mbuf, srcT, dstT, dv, ot, EE);
        agg_kernel<<<EE / 2, 256, 0, stream>>>(mbuf, srcL, dstL, dv, ot, EE);
        selfloop_kernel<<<NB, 256, 0, stream>>>(mbuf, dv, ot, NN * HH);
    }
}
